// MoERouter_48584670052581
// MI455X (gfx1250) — compile-verified, matrix operand order checked by reading, not run on hardware
//
#include <hip/hip_runtime.h>
#include <hip/hip_bf16.h>
#include <cstdint>

// ---------------- problem constants ----------------
#define B_DIM   4
#define L_DIM   4096
#define D_DIM   2048
#define E_DIM   64
#define N_TOK   (B_DIM * L_DIM)     // 16384 tokens
#define BT      128                 // tokens per workgroup
#define KC      32                  // K-chunk per WMMA step (bf16 K=32)
#define AUXW    0.01f
#define LDSTR   20                  // LDS row stride in dwords (padded, 16B-aligned, bank-conflict-free)

// ---------------- vector types (trivial, union-safe) ----------------
typedef __attribute__((ext_vector_type(16))) __bf16   v16bf;
typedef __attribute__((ext_vector_type(8)))  float    v8f;
typedef __attribute__((ext_vector_type(4)))  float    v4f;
typedef __attribute__((ext_vector_type(4)))  uint32_t v4u;

union FragAB { v16bf v; v4u u[2]; };
union FragC  { v8f   v; float f[8]; };

__device__ __forceinline__ uint32_t pack_bf16x2(float a, float b) {
  uint32_t ua = __float_as_uint(a);
  uint32_t ub = __float_as_uint(b);
  ua = (ua + 0x7FFFu + ((ua >> 16) & 1u)) >> 16;   // round-to-nearest-even
  ub = (ub + 0x7FFFu + ((ub >> 16) & 1u)) >> 16;
  return (ua & 0xFFFFu) | (ub << 16);
}

// ---------------- kernel 0: zero the per-expert accumulators ----------------
__global__ void moe_ws_init(float* ws) {
  ws[threadIdx.x] = 0.0f;   // 128 floats: [0..63]=top1 counts, [64..127]=prob sums
}

// ---------------- kernel 1: fused gate GEMM + top-2 softmax + aux partials ----------------
__global__ __launch_bounds__(256)
void moe_router_main(const float* __restrict__ x, const float* __restrict__ W,
                     float* __restrict__ out,
                     float* __restrict__ ws_cnt, float* __restrict__ ws_psum) {
  __shared__ uint32_t lds_x[BT][LDSTR];        // 128 rows of 32 bf16 (16 dwords used)
  __shared__ uint32_t lds_w[E_DIM][LDSTR];     // 64 rows of 32 bf16
  __shared__ float    lds_logits[BT][E_DIM + 1];
  __shared__ float    lds_cnt[E_DIM];
  __shared__ float    lds_psum[E_DIM];

  const int tid  = threadIdx.x;
  const int lane = tid & 31;           // wave32
  const int wv   = tid >> 5;           // wave id 0..7
  const int tile = blockIdx.x;         // 0..127 (128 tokens each)

  if (tid < E_DIM) { lds_cnt[tid] = 0.0f; lds_psum[tid] = 0.0f; }

  // cooperative staging assignments
  const int xrow = tid >> 1;                    // 0..127
  const int xcg  = (tid & 1) * 16;              // column group within 32
  const float* xptr = x + (size_t)(tile * BT + xrow) * D_DIM + xcg;
  const int wrow = tid >> 2;                    // 0..63
  const int wcg  = (tid & 3) * 8;
  const float* wptr = W + (size_t)wrow * D_DIM + wcg;

  // WMMA fragment addressing (ISA 16-bit A/B layout):
  // lane 0-15 hold K-octets {0..7} and {16..23}; lanes 16-31 hold {8..15},{24..31}
  const int hf   = lane >> 4;
  const int l16  = lane & 15;
  const int koffu = hf * 4;            // K byte-octet start, in dwords (8 bf16 = 4 dwords)
  const int mrow = wv * 16 + l16;      // A-matrix row for this lane

  FragC acc[4];
  #pragma unroll
  for (int e = 0; e < 4; ++e)
    #pragma unroll
    for (int r = 0; r < 8; ++r) acc[e].f[r] = 0.0f;

  for (int kc = 0; kc < D_DIM; kc += KC) {
    if (kc + KC < D_DIM) __builtin_prefetch(xptr + KC, 0, 3);   // global_prefetch next x chunk

    // ---- stage x tile: f32 -> packed bf16 in LDS ----
    v4f A0 = *(const v4f*)(xptr +  0);
    v4f A1 = *(const v4f*)(xptr +  4);
    v4f A2 = *(const v4f*)(xptr +  8);
    v4f A3 = *(const v4f*)(xptr + 12);
    v4u P0, P1;
    P0[0] = pack_bf16x2(A0[0], A0[1]); P0[1] = pack_bf16x2(A0[2], A0[3]);
    P0[2] = pack_bf16x2(A1[0], A1[1]); P0[3] = pack_bf16x2(A1[2], A1[3]);
    P1[0] = pack_bf16x2(A2[0], A2[1]); P1[1] = pack_bf16x2(A2[2], A2[3]);
    P1[2] = pack_bf16x2(A3[0], A3[1]); P1[3] = pack_bf16x2(A3[2], A3[3]);
    *(v4u*)&lds_x[xrow][(xcg >> 1) + 0] = P0;
    *(v4u*)&lds_x[xrow][(xcg >> 1) + 4] = P1;

    // ---- stage W tile ----
    v4f B0 = *(const v4f*)(wptr + 0);
    v4f B1 = *(const v4f*)(wptr + 4);
    v4u Q0;
    Q0[0] = pack_bf16x2(B0[0], B0[1]); Q0[1] = pack_bf16x2(B0[2], B0[3]);
    Q0[2] = pack_bf16x2(B1[0], B1[1]); Q0[3] = pack_bf16x2(B1[2], B1[3]);
    *(v4u*)&lds_w[wrow][wcg >> 1] = Q0;

    xptr += KC; wptr += KC;
    __syncthreads();

    // ---- A fragment: 16 tokens x 32 K ----
    FragAB a;
    a.u[0] = *(const v4u*)&lds_x[mrow][koffu];          // K elems {hf*8 .. +7}
    a.u[1] = *(const v4u*)&lds_x[mrow][8 + koffu];      // K elems {16+hf*8 .. +7}

    // ---- 4 expert tiles of 16 ----
    #pragma unroll
    for (int e = 0; e < 4; ++e) {
      FragAB b;
      const int erow = e * 16 + l16;
      b.u[0] = *(const v4u*)&lds_w[erow][koffu];
      b.u[1] = *(const v4u*)&lds_w[erow][8 + koffu];
      acc[e].v = __builtin_amdgcn_wmma_f32_16x16x32_bf16(
          false, a.v, false, b.v, (short)0, acc[e].v, false, false);
    }
    __syncthreads();
  }

  // ---- spill logits (C layout: lane<16 -> M=r, lane>=16 -> M=8+r; N=l16) ----
  #pragma unroll
  for (int e = 0; e < 4; ++e)
    #pragma unroll
    for (int r = 0; r < 8; ++r)
      lds_logits[wv * 16 + hf * 8 + r][e * 16 + l16] = acc[e].f[r];
  __syncthreads();

  // ---- per-token epilogue: top-2, renorm softmax, aux partials ----
  if (tid < BT) {
    const float* row = lds_logits[tid];
    float b1 = -3.4e38f; int i1 = 0;
    #pragma unroll 8
    for (int e = 0; e < E_DIM; ++e) { float v = row[e]; if (v > b1) { b1 = v; i1 = e; } }
    float b2 = -3.4e38f; int i2 = 0;
    #pragma unroll 8
    for (int e = 0; e < E_DIM; ++e) { float v = row[e]; if (e != i1 && v > b2) { b2 = v; i2 = e; } }

    float sum = 0.0f;
    #pragma unroll 8
    for (int e = 0; e < E_DIM; ++e) sum += __expf(row[e] - b1);
    float inv = 1.0f / sum;
    #pragma unroll 8
    for (int e = 0; e < E_DIM; ++e)
      atomicAdd(&lds_psum[e], __expf(row[e] - b1) * inv);   // ds_add_f32
    atomicAdd(&lds_cnt[i1], 1.0f);

    float e2 = __expf(b2 - b1);
    float w1 = 1.0f / (1.0f + e2);
    float w2 = e2 * w1;

    const int t = tile * BT + tid;
    out[2 * t]                  = (float)i1;       // top_k_indices (flattened, as f32)
    out[2 * t + 1]              = (float)i2;
    out[2 * N_TOK + 2 * t]      = w1;              // top_k_weights
    out[2 * N_TOK + 2 * t + 1]  = w2;
  }
  __syncthreads();

  if (tid < E_DIM) {
    atomicAdd(&ws_cnt[tid],  lds_cnt[tid]);
    atomicAdd(&ws_psum[tid], lds_psum[tid]);
  }
}

// ---------------- kernel 2: aux loss reduction ----------------
__global__ void moe_router_finalize(const float* __restrict__ ws_cnt,
                                    const float* __restrict__ ws_psum,
                                    float* __restrict__ out) {
  __shared__ float red[E_DIM];
  const int t = threadIdx.x;
  red[t] = ws_cnt[t] * ws_psum[t];
  __syncthreads();
  for (int s = E_DIM / 2; s > 0; s >>= 1) {
    if (t < s) red[t] += red[t + s];
    __syncthreads();
  }
  if (t == 0) {
    const float invN = 1.0f / (float)N_TOK;
    out[4 * N_TOK] = (float)E_DIM * AUXW * red[0] * invN * invN;
  }
}

// ---------------- host-side launcher ----------------
extern "C" void kernel_launch(void* const* d_in, const int* in_sizes, int n_in,
                              void* d_out, int out_size, void* d_ws, size_t ws_size,
                              hipStream_t stream) {
  const float* x = (const float*)d_in[0];   // [B, L, D] f32
  const float* W = (const float*)d_in[1];   // [E, D]    f32
  float* out = (float*)d_out;               // [2*N_TOK idx | 2*N_TOK wts | 1 aux]
  float* ws  = (float*)d_ws;                // [64 counts | 64 prob sums]

  moe_ws_init<<<1, 128, 0, stream>>>(ws);
  moe_router_main<<<N_TOK / BT, 256, 0, stream>>>(x, W, out, ws, ws + E_DIM);
  moe_router_finalize<<<1, E_DIM, 0, stream>>>(ws, ws + E_DIM, out);
}
